// NeighSuperpixelAgg_16398185136400
// MI455X (gfx1250) — matrix operand order repaired and checked by reading, not compiled, not run on hardware
//
#include <hip/hip_runtime.h>
#include <hip/hip_bf16.h>

typedef _Float16 v16h __attribute__((ext_vector_type(16)));
typedef _Float16 v8h  __attribute__((ext_vector_type(8)));
typedef _Float16 h2   __attribute__((ext_vector_type(2)));
typedef float    v8f  __attribute__((ext_vector_type(8)));
typedef float    f4   __attribute__((ext_vector_type(4)));
typedef unsigned int u32x4 __attribute__((ext_vector_type(4)));
typedef int          s32x4 __attribute__((ext_vector_type(4)));
typedef int          s32x8 __attribute__((ext_vector_type(8)));

#define DIM   256
#define HEADS 8
#define HW    128
#define KS    7
#define TILE  16
#define HALO  22                       // TILE + KS - 1
#define LDS_TILE_BYTES (HALO * HALO * DIM * 2)   // 247808 B, fits 320KB WGP LDS

#if defined(__has_builtin)
#if __has_builtin(__builtin_amdgcn_tensor_load_to_lds)
#define HAVE_TDM 1
#endif
#endif

// ---------------------------------------------------------------------------
// f32 -> f16 weight conversion (tiny)
// ---------------------------------------------------------------------------
__global__ void __launch_bounds__(256) f32_to_f16_kernel(const float* __restrict__ in,
                                                         _Float16* __restrict__ out, int n) {
  int i = blockIdx.x * blockDim.x + threadIdx.x;
  if (i < n) out[i] = (_Float16)in[i];
}

// ---------------------------------------------------------------------------
// V = f16(X @ W^T + b).  X: f32 [M,256].  W16: f16 [256,256] row-major.
// 8 waves/block, 32 M-rows/block, each wave computes a 16x64 tile.
// ---------------------------------------------------------------------------
__global__ void __launch_bounds__(256) gemm_v_kernel(const float* __restrict__ X,
                                                     const _Float16* __restrict__ W16,
                                                     const float* __restrict__ bias,
                                                     _Float16* __restrict__ V) {
  const int K    = DIM;
  const int wave = threadIdx.x >> 5;
  const int lane = threadIdx.x & 31;
  const int half = lane >> 4;
  const int l15  = lane & 15;
  const int n_base = (wave & 3) * 64;
  const long m_base = (long)blockIdx.x * 32 + (long)(wave >> 2) * 16;

  const float* arow = X + (m_base + l15) * K;

  v8f acc[4] = {v8f{}, v8f{}, v8f{}, v8f{}};

#pragma unroll
  for (int k0 = 0; k0 < K; k0 += 32) {
    f4 aa0 = *(const f4*)(arow + k0 + 8 * half);
    f4 aa1 = *(const f4*)(arow + k0 + 8 * half + 4);
    f4 aa2 = *(const f4*)(arow + k0 + 16 + 8 * half);
    f4 aa3 = *(const f4*)(arow + k0 + 16 + 8 * half + 4);
    v16h A;
#pragma unroll
    for (int i = 0; i < 4; ++i) {
      A[i]      = (_Float16)aa0[i];
      A[4 + i]  = (_Float16)aa1[i];
      A[8 + i]  = (_Float16)aa2[i];
      A[12 + i] = (_Float16)aa3[i];
    }
#pragma unroll
    for (int t = 0; t < 4; ++t) {
      const _Float16* brow = W16 + (n_base + t * 16 + l15) * K + k0;
      v8h blo = *(const v8h*)(brow + 8 * half);
      v8h bhi = *(const v8h*)(brow + 16 + 8 * half);
      v16h B = __builtin_shufflevector(blo, bhi, 0, 1, 2, 3, 4, 5, 6, 7,
                                       8, 9, 10, 11, 12, 13, 14, 15);
      acc[t] = __builtin_amdgcn_wmma_f32_16x16x32_f16(false, A, false, B,
                                                      (short)0, acc[t], false, false);
    }
  }

#pragma unroll
  for (int t = 0; t < 4; ++t) {
    const int n = n_base + t * 16 + l15;
    const float bn = bias[n];
#pragma unroll
    for (int r = 0; r < 8; ++r) {
      const long m = m_base + half * 8 + r;
      V[m * DIM + n] = (_Float16)(acc[t][r] + bn);
    }
  }
}

// ---------------------------------------------------------------------------
// Neighborhood aggregation with LDS-staged value halo.
// Block = 16x16 pixel tile of one image; halo = 22x22 pixels x 256ch f16
// (242KB) staged into LDS via the Tensor Data Mover (wave 0 issues one
// TENSOR_LOAD_TO_LDS, s_wait_tensorcnt, workgroup barrier), falling back to a
// cooperative vector copy if the TDM builtin is unavailable.
// Thread owns one half2 channel pair and 8x16 pixels; inner loop is
// ds_load_b32 + v_pk_fma_f16 per tap.
// ---------------------------------------------------------------------------
__global__ void __launch_bounds__(256) natt_agg_kernel(const _Float16* __restrict__ Vv,
                                                       const float* __restrict__ attn,
                                                       _Float16* __restrict__ Oo) {
  extern __shared__ _Float16 smem[];          // HALO*HALO*DIM f16, dynamic LDS @ offset 0
  h2* O2 = (h2*)Oo;
  const int b    = blockIdx.z;
  const int y0   = blockIdx.y * TILE;
  const int x0   = blockIdx.x * TILE;
  const int cp   = threadIdx.x & 127;         // half2 channel pair (c = 2*cp)
  const int ht   = threadIdx.x >> 7;          // upper/lower 8 rows of the tile
  const int head = cp >> 4;
  const int CP   = DIM / 2;                   // 128 half2 per pixel

  // halo origin, clamped so the 22x22 window is fully in-bounds
  int ybase = y0 - KS / 2; ybase = ybase < 0 ? 0 : (ybase > HW - HALO ? HW - HALO : ybase);
  int xbase = x0 - KS / 2; xbase = xbase < 0 ? 0 : (xbase > HW - HALO ? HW - HALO : xbase);

  const _Float16* gsrc = Vv + (size_t)((b * HW + ybase) * HW + xbase) * DIM;

#if HAVE_TDM
  if (threadIdx.x < 32) {                     // wave 0 issues the DMA
    unsigned long long ga = (unsigned long long)(const void*)gsrc;
    u32x4 g0;
    g0[0] = 1u;                               // count=1 (valid user descriptor)
    g0[1] = 0u;                               // lds_addr = 0 (start of dynamic LDS)
    g0[2] = (unsigned)(ga & 0xFFFFFFFFu);     // global_addr[31:0]
    g0[3] = (unsigned)((ga >> 32) & 0x01FFFFFFu) | (2u << 30);  // addr[56:32] | type=2
    s32x8 g1;
    g1[0] = (3 << 16);                        // workgroup_mask=0, data_size=3 (8B units)
    g1[1] = 0;                                // tensor_dim0 lo16 (dim0 = 1M units, lo=0)
    g1[2] = 0x10;                             // tensor_dim0 hi16=0x10 ; tensor_dim1 lo16=0
    g1[3] = 0x10 | (1408 << 16);              // tensor_dim1 hi16=0x10 ; tile_dim0=1408 (22px*512B/8)
    g1[4] = HALO;                             // tile_dim1=22 rows ; tile_dim2=0
    g1[5] = 8192;                             // tensor_dim0_stride = 64KiB row pitch / 8B
    g1[6] = 0;                                // stride hi / dim1_stride (unused for 2D)
    g1[7] = 0;
    s32x4 gz = {0, 0, 0, 0};                  // groups 2/3: zeros (2D tile)
#if __clang_major__ >= 23
    s32x8 z8 = {0, 0, 0, 0, 0, 0, 0, 0};
    __builtin_amdgcn_tensor_load_to_lds(g0, g1, gz, gz, z8, 0);
#else
    __builtin_amdgcn_tensor_load_to_lds(g0, g1, gz, gz, 0);
#endif
    __builtin_amdgcn_s_wait_tensorcnt(0);
  }
  __syncthreads();
#else
  // Fallback: cooperative 16B-chunk copy, global -> LDS
  {
    const int CHUNKS = HALO * HALO * (DIM * 2 / 16);      // 15488 uint4 chunks
    const int ROWCH  = HALO * (DIM * 2 / 16);             // 704 chunks per halo row
    for (int idx = threadIdx.x; idx < CHUNKS; idx += 256) {
      int row = idx / ROWCH;
      int off = idx - row * ROWCH;
      const uint4* g = (const uint4*)(Vv + (size_t)((b * HW + ybase + row) * HW + xbase) * DIM) + off;
      ((uint4*)smem)[idx] = *g;
    }
  }
  __syncthreads();
#endif

  const h2* vsh = (const h2*)smem;

  for (int r = 0; r < 8; ++r) {
    const int y = y0 + ht * 8 + r;
    int ys0 = y - KS / 2;
    ys0 = ys0 < 0 ? 0 : (ys0 > HW - KS ? HW - KS : ys0);
    const int li0 = ys0 - ybase;
    for (int xx = 0; xx < TILE; ++xx) {
      const int x = x0 + xx;
      int xs0 = x - KS / 2;
      xs0 = xs0 < 0 ? 0 : (xs0 > HW - KS ? HW - KS : xs0);
      const int lj0 = xs0 - xbase;
      const float* ap = attn + (size_t)(((b * HEADS + head) * HW + y) * HW + x) * (KS * KS);
      __builtin_prefetch(ap + KS * KS, 0, 0);
      h2 acc;
      acc[0] = (_Float16)0.f;
      acc[1] = (_Float16)0.f;
#pragma unroll
      for (int ki = 0; ki < KS; ++ki) {
        const h2* vrow = vsh + (size_t)((li0 + ki) * HALO + lj0) * CP + cp;
#pragma unroll
        for (int kj = 0; kj < KS; ++kj) {
          _Float16 w = (_Float16)ap[ki * KS + kj];
          h2 wv; wv[0] = w; wv[1] = w;
          acc += wv * vrow[kj * CP];            // ds_load_b32 + v_pk_fma_f16
        }
      }
      O2[(size_t)((b * HW + y) * HW + x) * CP + cp] = acc;
    }
  }
}

// ---------------------------------------------------------------------------
// Out = O @ Wp^T + bp.  O: f16 [M,256].  Output f32.
// ---------------------------------------------------------------------------
__global__ void __launch_bounds__(256) gemm_p_kernel(const _Float16* __restrict__ Amat,
                                                     const _Float16* __restrict__ W16,
                                                     const float* __restrict__ bias,
                                                     float* __restrict__ Out) {
  const int K    = DIM;
  const int wave = threadIdx.x >> 5;
  const int lane = threadIdx.x & 31;
  const int half = lane >> 4;
  const int l15  = lane & 15;
  const int n_base = (wave & 3) * 64;
  const long m_base = (long)blockIdx.x * 32 + (long)(wave >> 2) * 16;

  const _Float16* arow = Amat + (m_base + l15) * K;

  v8f acc[4] = {v8f{}, v8f{}, v8f{}, v8f{}};

#pragma unroll
  for (int k0 = 0; k0 < K; k0 += 32) {
    v8h lo = *(const v8h*)(arow + k0 + 8 * half);
    v8h hi = *(const v8h*)(arow + k0 + 16 + 8 * half);
    v16h A = __builtin_shufflevector(lo, hi, 0, 1, 2, 3, 4, 5, 6, 7,
                                     8, 9, 10, 11, 12, 13, 14, 15);
#pragma unroll
    for (int t = 0; t < 4; ++t) {
      const _Float16* brow = W16 + (n_base + t * 16 + l15) * K + k0;
      v8h blo = *(const v8h*)(brow + 8 * half);
      v8h bhi = *(const v8h*)(brow + 16 + 8 * half);
      v16h B = __builtin_shufflevector(blo, bhi, 0, 1, 2, 3, 4, 5, 6, 7,
                                       8, 9, 10, 11, 12, 13, 14, 15);
      acc[t] = __builtin_amdgcn_wmma_f32_16x16x32_f16(false, A, false, B,
                                                      (short)0, acc[t], false, false);
    }
  }

#pragma unroll
  for (int t = 0; t < 4; ++t) {
    const int n = n_base + t * 16 + l15;
    const float bn = bias[n];
#pragma unroll
    for (int r = 0; r < 8; ++r) {
      const long m = m_base + half * 8 + r;
      Out[m * DIM + n] = acc[t][r] + bn;
    }
  }
}

// ---------------------------------------------------------------------------
extern "C" void kernel_launch(void* const* d_in, const int* in_sizes, int n_in,
                              void* d_out, int out_size, void* d_ws, size_t ws_size,
                              hipStream_t stream) {
  (void)in_sizes; (void)n_in; (void)out_size; (void)ws_size;
  const float* x    = (const float*)d_in[0];  // [8,128,128,256]
  const float* attn = (const float*)d_in[1];  // [8,8,128,128,49]
  const float* Wv   = (const float*)d_in[2];  // [256,256]
  const float* bv   = (const float*)d_in[3];  // [256]
  const float* Wp   = (const float*)d_in[4];  // [256,256]
  const float* bp   = (const float*)d_in[5];  // [256]

  const size_t M = (size_t)8 * HW * HW;                 // 131072 pixels
  const size_t vbytes = M * DIM * sizeof(_Float16);     // 64 MiB

  char* ws = (char*)d_ws;
  _Float16* v16  = (_Float16*)ws;                       // f16 value tensor
  _Float16* o16  = (_Float16*)(ws + vbytes);            // f16 aggregated tensor
  _Float16* Wv16 = (_Float16*)(ws + 2 * vbytes);        // f16 weights
  _Float16* Wp16 = Wv16 + DIM * DIM;

  f32_to_f16_kernel<<<(DIM * DIM) / 256, 256, 0, stream>>>(Wv, Wv16, DIM * DIM);
  f32_to_f16_kernel<<<(DIM * DIM) / 256, 256, 0, stream>>>(Wp, Wp16, DIM * DIM);

  gemm_v_kernel<<<(int)(M / 32), 256, 0, stream>>>(x, Wv16, bv, v16);

  natt_agg_kernel<<<dim3(HW / TILE, HW / TILE, 8), 256, LDS_TILE_BYTES, stream>>>(v16, attn, o16);

  gemm_p_kernel<<<(int)(M / 32), 256, 0, stream>>>(o16, Wp16, bp, (float*)d_out);
}
